// MinLSTMCell_10393820856373
// MI455X (gfx1250) — matrix operand order repaired and checked by reading, not compiled, not run on hardware
//
#include <hip/hip_runtime.h>
#include <hip/hip_bf16.h>

// ---------------------------------------------------------------------------
// MinLSTM: pre = x @ W^T + b (M=32768, K=512, 3H=1536) fused with log-space
// gate math, then chunked Heinsen scan over L.
//
// Pipeline:
//   k1: convert x, W  fp32 -> bf16 (round-to-nearest) into ws
//   k2: WMMA bf16 GEMM; one wave = 64(M)x16(H) x 3 gates (12 accumulators).
//       B fragments loaded once per K-step and reused across 4 M-tiles
//       (~27 flop/L2-byte). Epilogue computes log_f, v = log_i + log g(h_pre).
//   k3: per-(channel,chunk) aggregate (a = sum log_f, b = local state)
//   k4: per-channel sequential combine of 64 chunk aggregates -> h_before
//   k5: per-(channel,chunk) replay, out = exp(log_h)
// ---------------------------------------------------------------------------

#define NB   8
#define LSEQ 4096
#define HID  512
#define KDIM 512
#define MROWS (NB * LSEQ)          // 32768
#define NCH  (NB * HID)            // 4096 channels
#define CHUNK 64
#define NCHUNK (LSEQ / CHUNK)      // 64
#define MT   4                     // M-tiles per wave

typedef __attribute__((ext_vector_type(16))) __bf16 v16bf;
typedef __attribute__((ext_vector_type(8)))  float  v8f;

union FragBF16 { uint4 q[2]; v16bf v; };

__device__ __forceinline__ float softplusf_(float x) {
    // stable: max(x,0) + log1p(exp(-|x|))
    return fmaxf(x, 0.0f) + log1pf(__expf(-fabsf(x)));
}

__device__ __forceinline__ float logaddexpf_(float x, float y) {
    float m = fmaxf(x, y);
    if (!(m > -INFINITY)) return -INFINITY;   // both -inf
    return m + log1pf(__expf(fminf(x, y) - m));
}

// -------------------------------- k1: fp32 -> bf16 -------------------------
__global__ __launch_bounds__(256)
void k1_convert(const float* __restrict__ x, const float* __restrict__ W,
                unsigned short* __restrict__ xbf, unsigned short* __restrict__ wbf) {
    const size_t NX = (size_t)MROWS * KDIM;          // 16,777,216
    const size_t NW = (size_t)3 * HID * KDIM;        //    786,432
    const size_t stride = (size_t)gridDim.x * blockDim.x;
    for (size_t i = (size_t)blockIdx.x * blockDim.x + threadIdx.x;
         i < NX + NW; i += stride) {
        float f;
        unsigned short* dst;
        if (i < NX) { f = x[i];      dst = xbf + i; }
        else        { f = W[i - NX]; dst = wbf + (i - NX); }
        unsigned int u = __float_as_uint(f);
        unsigned int r = (u + 0x7FFFu + ((u >> 16) & 1u)) >> 16;  // RNE
        *dst = (unsigned short)r;
    }
}

// --------------------- k2: WMMA GEMM + gate epilogue -----------------------
// grid: (MROWS/512, HID/16); block: 256 threads = 8 waves;
// wave = 4 M-tiles x 1 H-tile x 3 gates = 12 16x16 f32 accumulators.
__global__ __launch_bounds__(256)
void k2_gemm_gates(const unsigned short* __restrict__ xbf,
                   const unsigned short* __restrict__ wbf,
                   const float* __restrict__ bias,
                   float* __restrict__ lf_ws,
                   float* __restrict__ v_ws) {
    const int lane = threadIdx.x & 31;
    const int wave = threadIdx.x >> 5;
    const int m0   = (blockIdx.x * 8 + wave) * (16 * MT);
    const int h0   = blockIdx.y * 16;
    const int r    = lane & 15;     // A: row within tile; B/C: column within tile
    const int half = lane >> 4;

    v8f acc[MT][3];
#pragma unroll
    for (int t = 0; t < MT; ++t)
#pragma unroll
        for (int g = 0; g < 3; ++g)
            acc[t][g] = (v8f){};

    // A rows for the 4 M-tiles handled by this wave
    const unsigned short* arow0 = xbf + (size_t)(m0 + r) * KDIM;
    // B fragments: lane holds column (h0+r) of gate g, K-run k0 + 16*half
    const unsigned short* bF = wbf + (size_t)(h0 + r) * KDIM + half * 16;
    const unsigned short* bI = bF + (size_t)HID * KDIM;
    const unsigned short* bH = bI + (size_t)HID * KDIM;

#pragma unroll
    for (int k0 = 0; k0 < KDIM; k0 += 32) {
        // load the 3 gate-weight fragments once per K-step
        FragBF16 wb[3];
        wb[0].q[0] = *(const uint4*)(bF + k0);
        wb[0].q[1] = *(const uint4*)(bF + k0 + 8);
        wb[1].q[0] = *(const uint4*)(bI + k0);
        wb[1].q[1] = *(const uint4*)(bI + k0 + 8);
        wb[2].q[0] = *(const uint4*)(bH + k0);
        wb[2].q[1] = *(const uint4*)(bH + k0 + 8);

#pragma unroll
        for (int t = 0; t < MT; ++t) {
            const unsigned short* arow = arow0 + (size_t)(t * 16) * KDIM;
            FragBF16 a;
            a.q[0] = *(const uint4*)(arow + k0 + half * 16);
            a.q[1] = *(const uint4*)(arow + k0 + half * 16 + 8);

            acc[t][0] = __builtin_amdgcn_wmma_f32_16x16x32_bf16(
                false, a.v, false, wb[0].v, (short)0, acc[t][0], false, false);
            acc[t][1] = __builtin_amdgcn_wmma_f32_16x16x32_bf16(
                false, a.v, false, wb[1].v, (short)0, acc[t][1], false, false);
            acc[t][2] = __builtin_amdgcn_wmma_f32_16x16x32_bf16(
                false, a.v, false, wb[2].v, (short)0, acc[t][2], false, false);
        }
    }

    // C/D layout: element (vgpr i, lane l): M = mbase + i + 8*(l>>4), N = h0 + (l&15)
    const int h = h0 + r;
    const float biasF = bias[h];
    const float biasI = bias[HID + h];
    const float biasH = bias[2 * HID + h];

#pragma unroll
    for (int t = 0; t < MT; ++t) {
        const int mbase = m0 + t * 16;
#pragma unroll
        for (int i = 0; i < 8; ++i) {
            const int m = mbase + i + 8 * half;
            float fp = acc[t][0][i] + biasF;
            float ip = acc[t][1][i] + biasI;
            float hp = acc[t][2][i] + biasH;

            float diff   = softplusf_(-fp) - softplusf_(-ip);
            float log_f  = -softplusf_(diff);
            float log_i  = -softplusf_(-diff);
            float log_th = (hp >= 0.0f) ? __logf(hp + 0.5f) : -softplusf_(-hp);

            size_t idx = (size_t)m * HID + h;
            lf_ws[idx] = log_f;
            v_ws[idx]  = log_i + log_th;
        }
    }
}

// ----------------- k3: per-(channel, chunk) aggregates ---------------------
// state composition: apply (lf, v): a += lf; b = logaddexp(b + lf, v)
__global__ __launch_bounds__(256)
void k3_chunk_reduce(const float* __restrict__ lf_ws,
                     const float* __restrict__ v_ws,
                     float2* __restrict__ agg) {
    const int tid   = blockIdx.x * blockDim.x + threadIdx.x;  // 0 .. 262143
    const int ch    = tid & (NCH - 1);
    const int chunk = tid >> 12;
    const int n = ch >> 9, h = ch & (HID - 1);

    float a = 0.0f, b = -INFINITY;
    size_t idx = ((size_t)n * LSEQ + (size_t)chunk * CHUNK) * HID + h;
    for (int t = 0; t < CHUNK; ++t) {
        float lf = lf_ws[idx];
        float vv = v_ws[idx];
        a += lf;
        b = logaddexpf_(b + lf, vv);
        idx += HID;
    }
    agg[(size_t)chunk * NCH + ch] = make_float2(a, b);
}

// ----------------- k4: sequential carry scan over chunks -------------------
__global__ __launch_bounds__(256)
void k4_carry_scan(const float2* __restrict__ agg,
                   float* __restrict__ hbefore) {
    const int ch = blockIdx.x * blockDim.x + threadIdx.x;
    if (ch >= NCH) return;
    float hstate = -13.815510558f;   // log(1e-6)
    for (int c = 0; c < NCHUNK; ++c) {
        hbefore[(size_t)c * NCH + ch] = hstate;
        float2 ab = agg[(size_t)c * NCH + ch];
        hstate = logaddexpf_(hstate + ab.x, ab.y);
    }
}

// ----------------- k5: replay chunks, write exp(log_h) ---------------------
__global__ __launch_bounds__(256)
void k5_chunk_apply(const float* __restrict__ lf_ws,
                    const float* __restrict__ v_ws,
                    const float* __restrict__ hbefore,
                    float* __restrict__ out) {
    const int tid   = blockIdx.x * blockDim.x + threadIdx.x;
    const int ch    = tid & (NCH - 1);
    const int chunk = tid >> 12;
    const int n = ch >> 9, h = ch & (HID - 1);

    float hstate = hbefore[(size_t)chunk * NCH + ch];
    size_t idx = ((size_t)n * LSEQ + (size_t)chunk * CHUNK) * HID + h;
    for (int t = 0; t < CHUNK; ++t) {
        hstate = logaddexpf_(hstate + lf_ws[idx], v_ws[idx]);
        out[idx] = __expf(hstate);
        idx += HID;
    }
}

// ---------------------------------------------------------------------------
extern "C" void kernel_launch(void* const* d_in, const int* in_sizes, int n_in,
                              void* d_out, int out_size, void* d_ws, size_t ws_size,
                              hipStream_t stream) {
    (void)in_sizes; (void)n_in; (void)out_size; (void)ws_size;
    const float* x = (const float*)d_in[0];   // (8, 4096, 512)
    const float* W = (const float*)d_in[1];   // (1536, 512)
    const float* b = (const float*)d_in[2];   // (1536,)
    float* out = (float*)d_out;               // (8, 4096, 512)

    // workspace layout
    char* base = (char*)d_ws;
    const size_t xbf_bytes = (size_t)MROWS * KDIM * 2;          // 33,554,432
    const size_t wbf_bytes = (size_t)3 * HID * KDIM * 2;        //  1,572,864
    const size_t mat_bytes = (size_t)MROWS * HID * 4;           // 67,108,864
    unsigned short* xbf = (unsigned short*)base;
    unsigned short* wbf = (unsigned short*)(base + xbf_bytes);
    float* lf_ws = (float*)(base + xbf_bytes + wbf_bytes);
    float* v_ws  = (float*)(base + xbf_bytes + wbf_bytes + mat_bytes);
    float2* agg  = (float2*)(base + xbf_bytes + wbf_bytes + 2 * mat_bytes);
    float* hbefore = (float*)((char*)agg + (size_t)NCHUNK * NCH * sizeof(float2));

    // k1: convert inputs to bf16
    k1_convert<<<4096, 256, 0, stream>>>(x, W, xbf, wbf);

    // k2: fused WMMA GEMM + gate epilogue
    dim3 g2(MROWS / (8 * 16 * MT), HID / 16, 1);   // (64, 32)
    k2_gemm_gates<<<g2, 256, 0, stream>>>(xbf, wbf, b, lf_ws, v_ws);

    // k3..k5: chunked log-space scan
    k3_chunk_reduce<<<(NCH * NCHUNK) / 256, 256, 0, stream>>>(lf_ws, v_ws, agg);
    k4_carry_scan<<<NCH / 256, 256, 0, stream>>>(agg, hbefore);
    k5_chunk_apply<<<(NCH * NCHUNK) / 256, 256, 0, stream>>>(lf_ws, v_ws, hbefore, out);
}
